// MPNEncoder_33835752358355
// MI455X (gfx1250) — matrix operand order repaired and checked by reading, not compiled, not run on hardware
//
#include <hip/hip_runtime.h>
#include <hip/hip_bf16.h>

// ---------------- types ----------------
typedef __bf16 bf16;
typedef __attribute__((ext_vector_type(16))) __bf16 v16bf;
typedef __attribute__((ext_vector_type(8)))  __bf16 v8bf;
typedef __attribute__((ext_vector_type(8)))  float   v8f;
typedef __attribute__((__vector_size__(16))) int     v4i_vs;   // matches async-LDS builtin param

#define ASG __attribute__((address_space(1)))
#define ASS __attribute__((address_space(3)))

#define HID   300
#define NPAD  320           // padded hidden/output width (multiple of 64)
#define AFD   133           // atom feature dim
#define BFD   147           // bond feature dim
#define KPAD1 160           // ceil(147/32)*32
#define KPAD2 320           // ceil(300/32)*32
#define KPAD3 448           // ceil(433/32)*32
#define MAXNB 6

enum AMode { A_F32 = 0, A_BF16 = 1, A_GATHER = 2 };
enum Epi   { E_INP_MSG = 0, E_ADD_INP_RELU = 1, E_BIAS_RELU_F32 = 2 };

// ---------------- weight transpose + pad to bf16 [NPAD][KPAD] ----------------
__global__ __launch_bounds__(256) void k_transpose_pad(const float* __restrict__ W,
                                                       bf16* __restrict__ Wt,
                                                       int K, int N, int KPAD) {
  long t = (long)blockIdx.x * 256 + threadIdx.x;
  long total = (long)NPAD * KPAD;
  if (t >= total) return;
  int n = (int)(t / KPAD), k = (int)(t % KPAD);
  float v = (k < K && n < N) ? W[(long)k * N + n] : 0.0f;
  Wt[t] = (bf16)v;
}

__global__ __launch_bounds__(256) void k_pad_bias(const float* __restrict__ b,
                                                  float* __restrict__ bp) {
  int t = blockIdx.x * 256 + threadIdx.x;
  if (t < NPAD) bp[t] = (t < HID) ? b[t] : 0.0f;
}

// ---------------- tiled WMMA GEMM ----------------
// One wave per 16-row M tile.  A fragments (full K) live in registers for the
// whole N sweep; B (pre-transposed weights, [NPAD][KPAD] bf16) is streamed
// through LDS in 64-row slabs via GLOBAL_LOAD_ASYNC_TO_LDS (ASYNCcnt path).
template <int KPAD, int AMODE, int EPI>
__global__ __launch_bounds__(256) void k_gemm(
    const void* __restrict__ Aptr,        // A_F32: f32 [M][147]; A_BF16: bf16 [M][KPAD]
    const bf16* __restrict__ Wt,          // [NPAD][KPAD] bf16 (B transposed)
    const bf16* __restrict__ inp,         // E_ADD_INP_RELU addend, bf16 [M][NPAD]
    const float* __restrict__ bias,       // E_BIAS_RELU_F32 bias, f32 [NPAD]
    bf16* __restrict__ out_bf0,           // E_INP_MSG: inp-out ; E_ADD: msg_next
    bf16* __restrict__ out_bf1,           // E_INP_MSG: msg-out
    float* __restrict__ out_f32,          // E_BIAS_RELU_F32 output, [M][NPAD]
    const int* __restrict__ b2a, const int* __restrict__ b2revb,   // A_GATHER
    const bf16* __restrict__ amsg, const bf16* __restrict__ msgsrc,// A_GATHER srcs
    int M) {
  constexpr int KT = KPAD / 32;           // k-tiles of 32
  __shared__ bf16 ldsB[64 * KPAD];        // one 64-row weight slab (<= 57 KB)

  const int wave  = threadIdx.x >> 5;
  const int lane  = threadIdx.x & 31;
  const int lane16 = lane & 15;
  const int sel   = lane >> 4;            // half-wave select (K-chunk select)
  const int mtile = blockIdx.x * 8 + wave;
  const bool active = (mtile * 16) < M;
  const long row  = (long)mtile * 16 + lane16;

  // ---- load A fragments for full K into registers ----
  v16bf afrag[KT];
  if (active) {
    if (AMODE == A_F32) {
      const float* Arow = ((const float*)Aptr) + row * BFD;
      for (int kt = 0; kt < KT; ++kt) {
        v8bf lo, hi;
        int kb = kt * 32 + sel * 8;
        for (int i = 0; i < 8; ++i) {
          int k0 = kb + i, k1 = kb + 16 + i;
          lo[i] = (bf16)((k0 < BFD) ? Arow[k0] : 0.0f);
          hi[i] = (bf16)((k1 < BFD) ? Arow[k1] : 0.0f);
        }
        afrag[kt] = __builtin_shufflevector(lo, hi, 0,1,2,3,4,5,6,7,8,9,10,11,12,13,14,15);
      }
    } else if (AMODE == A_BF16) {
      const bf16* Arow = ((const bf16*)Aptr) + row * KPAD;
      for (int kt = 0; kt < KT; ++kt) {
        v8bf lo = *(const v8bf*)(Arow + kt * 32 + sel * 8);
        v8bf hi = *(const v8bf*)(Arow + kt * 32 + sel * 8 + 16);
        afrag[kt] = __builtin_shufflevector(lo, hi, 0,1,2,3,4,5,6,7,8,9,10,11,12,13,14,15);
      }
    } else { // A_GATHER: a_message[b2a[row]] - message[b2revb[row]]
      const bf16* ra = amsg   + (long)b2a[row]   * NPAD;
      const bf16* rr = msgsrc + (long)b2revb[row] * NPAD;
      for (int kt = 0; kt < KT; ++kt) {
        int kb = kt * 32 + sel * 8;
        v8bf alo = *(const v8bf*)(ra + kb),      rlo = *(const v8bf*)(rr + kb);
        v8bf ahi = *(const v8bf*)(ra + kb + 16), rhi = *(const v8bf*)(rr + kb + 16);
        v8bf lo, hi;
        for (int i = 0; i < 8; ++i) {
          lo[i] = (bf16)((float)alo[i] - (float)rlo[i]);
          hi[i] = (bf16)((float)ahi[i] - (float)rhi[i]);
        }
        afrag[kt] = __builtin_shufflevector(lo, hi, 0,1,2,3,4,5,6,7,8,9,10,11,12,13,14,15);
      }
    }
  }

  // ---- sweep N in 64-row weight slabs staged through LDS ----
  for (int nsup = 0; nsup < NPAD / 64; ++nsup) {
    __syncthreads();
    {
      const uint4* src = (const uint4*)(Wt + (long)nsup * 64 * KPAD);
      uint4* dst = (uint4*)ldsB;
      const int chunks = (64 * KPAD * 2) / 16;   // 16B chunks
#if defined(__gfx1250__) && __has_builtin(__builtin_amdgcn_global_load_async_to_lds_b128)
      // direct global->LDS DMA path, tracked by ASYNCcnt (no VGPR round-trip)
      for (int i = threadIdx.x; i < chunks; i += 256)
        __builtin_amdgcn_global_load_async_to_lds_b128(
            (ASG v4i_vs*)(src + i), (ASS v4i_vs*)(dst + i), 0, 0);
#if __has_builtin(__builtin_amdgcn_s_wait_asynccnt)
      __builtin_amdgcn_s_wait_asynccnt(0);
#else
      asm volatile("s_wait_asynccnt 0" ::: "memory");
#endif
#else
      for (int i = threadIdx.x; i < chunks; i += 256) dst[i] = src[i];
#endif
      if (nsup + 1 < NPAD / 64)                  // hint next slab -> global_prefetch
        __builtin_prefetch(Wt + (long)(nsup + 1) * 64 * KPAD + threadIdx.x * 64, 0, 1);
    }
    __syncthreads();
    if (!active) continue;

    for (int nt = 0; nt < 4; ++nt) {
      v8f acc = {};
      const bf16* brow = ldsB + (nt * 16 + lane16) * KPAD + sel * 8;
      for (int kt = 0; kt < KT; ++kt) {
        v8bf blo = *(const v8bf*)(brow + kt * 32);
        v8bf bhi = *(const v8bf*)(brow + kt * 32 + 16);
        v16bf b = __builtin_shufflevector(blo, bhi, 0,1,2,3,4,5,6,7,8,9,10,11,12,13,14,15);
        acc = __builtin_amdgcn_wmma_f32_16x16x32_bf16(
            false, afrag[kt], false, b, (short)0, acc, false, false);
      }
      // ---- epilogue: C/D layout = VGPR i -> row (i + sel*8), col = lane16 ----
      const int  ncol  = nsup * 64 + nt * 16 + lane16;
      const long rbase = (long)mtile * 16 + sel * 8;
      for (int i = 0; i < 8; ++i) {
        long r = rbase + i;
        float v = acc[i];
        if (EPI == E_INP_MSG) {
          out_bf0[r * NPAD + ncol] = (bf16)v;                 // inp (pre-activation)
          out_bf1[r * NPAD + ncol] = (bf16)fmaxf(v, 0.0f);    // message = relu(inp)
        } else if (EPI == E_ADD_INP_RELU) {
          float s = v + (float)inp[r * NPAD + ncol];
          out_bf0[r * NPAD + ncol] = (bf16)fmaxf(s, 0.0f);
        } else {
          float s = v + bias[ncol];
          out_f32[r * NPAD + ncol] = fmaxf(s, 0.0f);
        }
      }
    }
  }
}

// ---------------- neighbor gather-sum: amsg[a] = sum_j msg[a2b[a][j]] ----------------
__global__ __launch_bounds__(256) void k_gather_sum(const bf16* __restrict__ msg,
                                                    const int* __restrict__ a2b,
                                                    bf16* __restrict__ amsg, int n_atoms) {
  long t = (long)blockIdx.x * 256 + threadIdx.x;   // over n_atoms * (NPAD/8)
  if (t >= (long)n_atoms * (NPAD / 8)) return;
  int a  = (int)(t / (NPAD / 8));
  int c8 = (int)(t % (NPAD / 8));
  float s[8] = {0, 0, 0, 0, 0, 0, 0, 0};
  for (int j = 0; j < MAXNB; ++j) {
    long b = a2b[(long)a * MAXNB + j];
    v8bf v = *(const v8bf*)(msg + b * NPAD + c8 * 8);
    for (int i = 0; i < 8; ++i) s[i] += (float)v[i];
  }
  v8bf o;
  for (int i = 0; i < 8; ++i) o[i] = (bf16)s[i];
  *(v8bf*)(amsg + (long)a * NPAD + c8 * 8) = o;
}

// --------- build concat input [f_atoms | gather-sum(msg)] padded to KPAD3 ---------
__global__ __launch_bounds__(256) void k_build_ainput(const float* __restrict__ f_atoms,
                                                      const bf16* __restrict__ msg,
                                                      const int* __restrict__ a2b,
                                                      bf16* __restrict__ a_input, int n_atoms) {
  long t = (long)blockIdx.x * 256 + threadIdx.x;   // over n_atoms * KPAD3
  if (t >= (long)n_atoms * KPAD3) return;
  int a = (int)(t / KPAD3), c = (int)(t % KPAD3);
  float v = 0.0f;
  if (c < AFD) {
    v = f_atoms[(long)a * AFD + c];
  } else if (c < AFD + HID) {
    int cc = c - AFD;
    for (int j = 0; j < MAXNB; ++j)
      v += (float)msg[(long)a2b[(long)a * MAXNB + j] * NPAD + cc];
  }
  a_input[t] = (bf16)v;
}

// ---------------- per-molecule mean pool ----------------
__global__ __launch_bounds__(256) void k_pool(const float* __restrict__ hidden,
                                              float* __restrict__ out, int n_mols, int apm) {
  long t = (long)blockIdx.x * 256 + threadIdx.x;   // over n_mols * HID
  if (t >= (long)n_mols * HID) return;
  int m = (int)(t / HID), h = (int)(t % HID);
  float s = 0.0f;
  for (int i = 0; i < 20; ++i)
    if (i < apm) s += hidden[((long)m * apm + i) * NPAD + h];
  out[t] = s / (float)apm;
}

// ---------------- driver ----------------
extern "C" void kernel_launch(void* const* d_in, const int* in_sizes, int n_in,
                              void* d_out, int out_size, void* d_ws, size_t ws_size,
                              hipStream_t stream) {
  const float* f_atoms = (const float*)d_in[0];
  const float* f_bonds = (const float*)d_in[1];
  const int*   a2b     = (const int*)d_in[2];
  const int*   b2a     = (const int*)d_in[3];
  const int*   b2revb  = (const int*)d_in[4];
  const float* W_i     = (const float*)d_in[5];
  const float* W_h     = (const float*)d_in[6];
  const float* W_o     = (const float*)d_in[7];
  const float* b_o     = (const float*)d_in[8];

  const int n_atoms = in_sizes[0] / AFD;   // 200000
  const int n_bonds = in_sizes[1] / BFD;   // 400000
  const int n_mols  = out_size / HID;      // 10000
  const int apm     = n_atoms / n_mols;    // 20
  float* out = (float*)d_out;

  // carve workspace (256B aligned)
  size_t off = 0;
  auto take = [&](size_t bytes) {
    off = (off + 255) & ~(size_t)255;
    size_t o = off; off += bytes; return o;
  };
  char* ws = (char*)d_ws;
  bf16*  Wt_i   = (bf16*)(ws + take((size_t)NPAD * KPAD1 * sizeof(bf16)));
  bf16*  Wt_h   = (bf16*)(ws + take((size_t)NPAD * KPAD2 * sizeof(bf16)));
  bf16*  Wt_o   = (bf16*)(ws + take((size_t)NPAD * KPAD3 * sizeof(bf16)));
  float* biasP  = (float*)(ws + take((size_t)NPAD * sizeof(float)));
  bf16*  inpB   = (bf16*)(ws + take((size_t)n_bonds * NPAD * sizeof(bf16)));
  bf16*  msg0   = (bf16*)(ws + take((size_t)n_bonds * NPAD * sizeof(bf16)));
  bf16*  msg1   = (bf16*)(ws + take((size_t)n_bonds * NPAD * sizeof(bf16)));
  bf16*  amsg   = (bf16*)(ws + take((size_t)n_atoms * NPAD * sizeof(bf16)));
  bf16*  ainp   = (bf16*)(ws + take((size_t)n_atoms * KPAD3 * sizeof(bf16)));
  float* hidden = (float*)(ws + take((size_t)n_atoms * NPAD * sizeof(float)));
  (void)ws_size; (void)n_in;

  auto cdiv = [](long a, long b) { return (unsigned)((a + b - 1) / b); };

  // 1) weight prep
  k_transpose_pad<<<cdiv((long)NPAD * KPAD1, 256), 256, 0, stream>>>(W_i, Wt_i, BFD, HID, KPAD1);
  k_transpose_pad<<<cdiv((long)NPAD * KPAD2, 256), 256, 0, stream>>>(W_h, Wt_h, HID, HID, KPAD2);
  k_transpose_pad<<<cdiv((long)NPAD * KPAD3, 256), 256, 0, stream>>>(W_o, Wt_o, AFD + HID, HID, KPAD3);
  k_pad_bias<<<cdiv(NPAD, 256), 256, 0, stream>>>(b_o, biasP);

  // 2) inp = f_bonds @ W_i ; msg0 = relu(inp)
  k_gemm<KPAD1, A_F32, E_INP_MSG><<<cdiv(n_bonds, 128), 256, 0, stream>>>(
      f_bonds, Wt_i, nullptr, nullptr, inpB, msg0, nullptr,
      nullptr, nullptr, nullptr, nullptr, n_bonds);

  // 3) DEPTH-1 = 2 message-passing steps (fused gather-sub inside GEMM A-load)
  bf16* cur = msg0; bf16* nxt = msg1;
  for (int d = 0; d < 2; ++d) {
    k_gather_sum<<<cdiv((long)n_atoms * (NPAD / 8), 256), 256, 0, stream>>>(cur, a2b, amsg, n_atoms);
    k_gemm<KPAD2, A_GATHER, E_ADD_INP_RELU><<<cdiv(n_bonds, 128), 256, 0, stream>>>(
        nullptr, Wt_h, inpB, nullptr, nxt, nullptr, nullptr,
        b2a, b2revb, amsg, cur, n_bonds);
    bf16* t = cur; cur = nxt; nxt = t;
  }

  // 4) readout: a_input = [f_atoms | gather-sum(cur)], hidden = relu(a_input@W_o + b_o)
  k_build_ainput<<<cdiv((long)n_atoms * KPAD3, 256), 256, 0, stream>>>(f_atoms, cur, a2b, ainp, n_atoms);
  k_gemm<KPAD3, A_BF16, E_BIAS_RELU_F32><<<cdiv(n_atoms, 128), 256, 0, stream>>>(
      ainp, Wt_o, nullptr, biasP, nullptr, nullptr, hidden,
      nullptr, nullptr, nullptr, nullptr, n_atoms);

  // 5) per-molecule mean pool
  k_pool<<<cdiv((long)n_mols * HID, 256), 256, 0, stream>>>(hidden, out, n_mols, apm);
}
